// TransformerNetwork_53755810676977
// MI455X (gfx1250) — matrix operand order repaired and checked by reading, compiled, not run on hardware
//
#include <hip/hip_runtime.h>

typedef __attribute__((ext_vector_type(16))) __bf16 v16bf;
typedef __attribute__((ext_vector_type(8)))  __bf16 v8bf;
typedef __attribute__((ext_vector_type(8)))  float  v8f;

#define Bc   4
#define SEQL 1024
#define Hd   512
#define NHd  8
#define DHd  64
#define Fd   2048
#define Ed   128
#define Ld   6
#define VDd  32000
#define BSr  (Bc * SEQL)

// ---- float -> bf16 (round to nearest even) helpers ----
static __device__ __forceinline__ unsigned short bfbits(float f) {
  unsigned u = __builtin_bit_cast(unsigned, f);
  u += 0x7FFFu + ((u >> 16) & 1u);
  return (unsigned short)(u >> 16);
}
static __device__ __forceinline__ __bf16 f2bf(float f) {
  return __builtin_bit_cast(__bf16, bfbits(f));
}
static __device__ __forceinline__ unsigned pack_bf2(float lo, float hi) {
  return (unsigned)bfbits(lo) | ((unsigned)bfbits(hi) << 16);
}
static __device__ __forceinline__ void put4(v16bf& f, int base, float4 x) {
  f[base + 0] = f2bf(x.x);
  f[base + 1] = f2bf(x.y);
  f[base + 2] = f2bf(x.z);
  f[base + 3] = f2bf(x.w);
}

static __device__ __forceinline__ v8f wmma_bf16(v16bf a, v16bf b, v8f c) {
  return __builtin_amdgcn_wmma_f32_16x16x32_bf16(false, a, false, b, (short)0, c,
                                                 false, false);
}

// ---- CDNA5 split workgroup barrier: LDS-release + arrive, and wait ----
// (mnemonics identical to what the backend emits for gfx1250)
static __device__ __forceinline__ void wg_signal_lds() {
  asm volatile("s_wait_dscnt 0x0\n\ts_barrier_signal -1" ::: "memory");
}
static __device__ __forceinline__ void wg_wait() {
  asm volatile("s_barrier_wait -1" ::: "memory");
}
static __device__ __forceinline__ void lds_wait() {
  asm volatile("s_wait_dscnt 0x0" ::: "memory");
}

// -------------------- embedding gather (float4) --------------------
__global__ __launch_bounds__(32) void gather_kernel(const int* __restrict__ idx,
                                                    const float4* __restrict__ emb,
                                                    float4* __restrict__ out) {
  int row = blockIdx.x;
  int t = threadIdx.x;  // Ed/4 == 32
  out[(size_t)row * 32 + t] = emb[(size_t)idx[row] * 32 + t];
}

// -------------------- positional add (float4): out = x + pos[s,h] ----------
__global__ __launch_bounds__(256) void add_pos_kernel(const float4* __restrict__ x,
                                                      const float4* __restrict__ pos,
                                                      float4* __restrict__ out,
                                                      int total4, int shmask4) {
  int i = blockIdx.x * blockDim.x + threadIdx.x;
  if (i < total4) {
    float4 a = x[i];
    float4 p = pos[i & shmask4];
    out[i] = make_float4(a.x + p.x, a.y + p.y, a.z + p.z, a.w + p.w);
  }
}

// -------------------- tiled GEMM: bf16 WMMA, fp32 accumulate --------------------
// Block = 256 threads (8 waves) computes a 128(M) x 64(N) tile, K-step 32.
// Software-pipelined, double-buffered LDS, explicit split barriers so the next
// panel's global loads are issued between barrier-signal and barrier-wait and
// stay in flight through the whole compute phase.
__global__ __launch_bounds__(256) void gemm_bf16_wmma_kernel(
    const float* __restrict__ A, const float* __restrict__ W,
    float* __restrict__ C, int M, int N, int K,
    const float* __restrict__ bias, int relu) {
  __shared__ __align__(16) __bf16 sA[2][128 * 32];
  __shared__ __align__(16) __bf16 sB[2][64 * 32];
  const int t = threadIdx.x;
  const int lane = t & 31;
  const int w = t >> 5;      // wave 0..7
  const int wn = w & 1;      // n sub-block (2 x 32 cols)
  const int wm = w >> 1;     // m sub-block (4 x 32 rows)
  const int half = lane >> 4, lm = lane & 15;
  const int tn = blockIdx.x * 64;
  const int tm = blockIdx.y * 128;

  v8f acc[2][2] = {};
  float4 ra[4], rb[2];  // staging registers for the in-flight panel

  auto load_panels = [&](int kk) {
#pragma unroll
    for (int i = 0; i < 4; ++i) {
      int idx = t * 4 + i;  // linear over [row][k4]
      int row = idx >> 3, k4 = idx & 7;
      const float* ap = A + (size_t)(tm + row) * K + kk + k4 * 4;
      ra[i] = *(const float4*)ap;
      if (kk + 32 < K) __builtin_prefetch(ap + 32, 0, 1);
    }
#pragma unroll
    for (int i = 0; i < 2; ++i) {
      int idx = t * 2 + i;  // linear over [k][n4]
      int kr = idx >> 4, n4 = idx & 15;
      const float* wp = W + (size_t)(kk + kr) * N + tn + n4 * 4;
      rb[i] = *(const float4*)wp;
      if (kk + 32 < K) __builtin_prefetch(wp + (size_t)32 * N, 0, 1);
    }
  };
  auto store_panels = [&](int buf) {
#pragma unroll
    for (int i = 0; i < 4; ++i) {
      int idx = t * 4 + i;
      int row = idx >> 3, k4 = idx & 7;
      ((uint2*)sA[buf])[row * 8 + k4] =
          make_uint2(pack_bf2(ra[i].x, ra[i].y), pack_bf2(ra[i].z, ra[i].w));
    }
#pragma unroll
    for (int i = 0; i < 2; ++i) {
      int idx = t * 2 + i;
      int kr = idx >> 4, n4 = idx & 15;
      __bf16* bp = sB[buf];
      bp[(n4 * 4 + 0) * 32 + kr] = f2bf(rb[i].x);
      bp[(n4 * 4 + 1) * 32 + kr] = f2bf(rb[i].y);
      bp[(n4 * 4 + 2) * 32 + kr] = f2bf(rb[i].z);
      bp[(n4 * 4 + 3) * 32 + kr] = f2bf(rb[i].w);
    }
  };
  auto compute = [&](int buf) {
    v16bf aF[2], bF[2];
#pragma unroll
    for (int mi = 0; mi < 2; ++mi) {
      int row = wm * 32 + mi * 16 + lm;
      const v8bf* pa = (const v8bf*)(sA[buf] + row * 32 + half * 8);
      aF[mi] = __builtin_shufflevector(pa[0], pa[2], 0, 1, 2, 3, 4, 5, 6, 7, 8, 9,
                                       10, 11, 12, 13, 14, 15);
    }
#pragma unroll
    for (int ni = 0; ni < 2; ++ni) {
      int col = wn * 32 + ni * 16 + lm;
      const v8bf* pb = (const v8bf*)(sB[buf] + col * 32 + half * 8);
      bF[ni] = __builtin_shufflevector(pb[0], pb[2], 0, 1, 2, 3, 4, 5, 6, 7, 8, 9,
                                       10, 11, 12, 13, 14, 15);
    }
#pragma unroll
    for (int mi = 0; mi < 2; ++mi)
#pragma unroll
      for (int ni = 0; ni < 2; ++ni)
        acc[mi][ni] = wmma_bf16(aF[mi], bF[ni], acc[mi][ni]);
  };

  const int P = K >> 5;  // number of 32-wide K panels
  // prologue: panel 0 into LDS0; panel 1's loads issued inside the barrier gap
  load_panels(0);
  store_panels(0);
  wg_signal_lds();
  if (P > 1) load_panels(32);
  wg_wait();
  int buf = 0;
  for (int p = 1; p < P; ++p) {
    compute(buf);            // panel p-1 from LDS[buf]
    store_panels(buf ^ 1);   // panel p (regs loaded one iteration ago)
    wg_signal_lds();         // dscnt-only release + arrive
    if (p + 1 < P) load_panels((p + 1) << 5);  // loads fly across the barrier
    wg_wait();
    buf ^= 1;
  }
  compute(buf);

  // ---- epilogue: hoisted bias, optional ReLU ----
#pragma unroll
  for (int ni = 0; ni < 2; ++ni) {
    int col = tn + wn * 32 + ni * 16 + lm;
    float bv = bias ? bias[col] : 0.0f;
#pragma unroll
    for (int mi = 0; mi < 2; ++mi) {
      int rowb = tm + wm * 32 + mi * 16 + half * 8;
#pragma unroll
      for (int v = 0; v < 8; ++v) {
        float val = acc[mi][ni][v] + bv;
        if (relu) val = fmaxf(val, 0.0f);
        C[(size_t)(rowb + v) * N + col] = val;
      }
    }
  }
}

// -------------------- fused residual LayerNorm: out = res + ln(x)*scale + bias ----
__global__ __launch_bounds__(256) void ln_add_kernel(
    const float* __restrict__ x, const float* __restrict__ res,
    const float* __restrict__ bias, const float* __restrict__ scale,
    float* __restrict__ out) {
  __shared__ float shs[8], shq[8];
  const int row = blockIdx.x, t = threadIdx.x;
  const float* xr = x + (size_t)row * Hd;
  float v0 = xr[t], v1 = xr[t + 256];
  float s = v0 + v1, q = v0 * v0 + v1 * v1;
#pragma unroll
  for (int m = 16; m; m >>= 1) {
    s += __shfl_xor(s, m, 32);
    q += __shfl_xor(q, m, 32);
  }
  if ((t & 31) == 0) { shs[t >> 5] = s; shq[t >> 5] = q; }
  __syncthreads();
  float S = 0.0f, Q = 0.0f;
#pragma unroll
  for (int i = 0; i < 8; ++i) { S += shs[i]; Q += shq[i]; }
  float mu = S * (1.0f / Hd);
  float var = Q * (1.0f / Hd) - mu * mu;
  float rs = rsqrtf(var + 1e-6f);
  const float* rr = res + (size_t)row * Hd;
  out[(size_t)row * Hd + t] = rr[t] + (v0 - mu) * rs * scale[t] + bias[t];
  out[(size_t)row * Hd + t + 256] =
      rr[t + 256] + (v1 - mu) * rs * scale[t + 256] + bias[t + 256];
}

// -------------------- flash attention (online softmax), one wave / 16-q tile ------
// Q,K,V,O: [B, S, NH*DH] fp32. S = (Q K^T)/8 (+causal mask), O = softmax(S) V.
__global__ __launch_bounds__(32) void attn_fa_wmma_kernel(
    const float* __restrict__ Q, const float* __restrict__ K,
    const float* __restrict__ V, float* __restrict__ O,
    int SQ, int SK, int causal) {
  __shared__ __align__(16) float sP[16 * 32];
  const int lane = threadIdx.x;
  const int half = lane >> 4, lm = lane & 15;
  const int qt = blockIdx.x;
  const int head = blockIdx.y;
  const int b = blockIdx.z;

  // Q tile (16 x 64) as two A fragments; lane's elements are two 16B runs.
  v16bf qA[2];
  const float* Qrow = Q + ((size_t)(b * SQ + qt * 16 + lm) * Hd) + head * DHd;
#pragma unroll
  for (int kk = 0; kk < 2; ++kk) {
    const float4* qa = (const float4*)(Qrow + kk * 32 + half * 8);
    const float4* qb = (const float4*)(Qrow + kk * 32 + 16 + half * 8);
    put4(qA[kk], 0, qa[0]);
    put4(qA[kk], 4, qa[1]);
    put4(qA[kk], 8, qb[0]);
    put4(qA[kk], 12, qb[1]);
  }

  float m_s[8], l_s[8];
  v8f oacc[4] = {};
#pragma unroll
  for (int v = 0; v < 8; ++v) { m_s[v] = -1e30f; l_s[v] = 0.0f; }

  int nc = SK / 32;
  if (causal) nc = (qt * 16 + 15) / 32 + 1;

  for (int c = 0; c < nc; ++c) {
    // ---- S tile: Q (16x64) x K^T (64x32) -> two 16x16 f32 frags ----
    v8f s0 = {}, s1 = {};
#pragma unroll
    for (int kk = 0; kk < 2; ++kk) {
      // B frag: lane's column is key (fixed), K-rows run along contiguous d.
      const float* Kr0 =
          K + (size_t)(b * SK + c * 32 + lm) * Hd + head * DHd + kk * 32 + half * 8;
      const float* Kr1 = Kr0 + (size_t)16 * Hd;  // keys +16
      const float4* k40 = (const float4*)Kr0;
      const float4* k41 = (const float4*)Kr1;
      v16bf kb0, kb1;
      put4(kb0, 0, k40[0]); put4(kb0, 4, k40[1]);
      put4(kb0, 8, k40[4]); put4(kb0, 12, k40[5]);
      put4(kb1, 0, k41[0]); put4(kb1, 4, k41[1]);
      put4(kb1, 8, k41[4]); put4(kb1, 12, k41[5]);
      s0 = wmma_bf16(qA[kk], kb0, s0);
      s1 = wmma_bf16(qA[kk], kb1, s1);
    }

    // ---- scale, mask, online softmax ----
    float p0[8], p1[8], rm[8];
#pragma unroll
    for (int v = 0; v < 8; ++v) {
      float a0 = s0[v] * 0.125f;
      float a1 = s1[v] * 0.125f;
      if (causal) {
        int qrow = qt * 16 + v + (half << 3);
        if (c * 32 + lm > qrow) a0 += -1e9f;
        if (c * 32 + 16 + lm > qrow) a1 += -1e9f;
      }
      s0[v] = a0;
      s1[v] = a1;
      float r = fmaxf(a0, a1);
      r = fmaxf(r, __shfl_xor(r, 1, 32));
      r = fmaxf(r, __shfl_xor(r, 2, 32));
      r = fmaxf(r, __shfl_xor(r, 4, 32));
      r = fmaxf(r, __shfl_xor(r, 8, 32));
      rm[v] = r;
    }
#pragma unroll
    for (int v = 0; v < 8; ++v) {
      float mn = fmaxf(m_s[v], rm[v]);
      float alpha = __expf(m_s[v] - mn);
      p0[v] = __expf(s0[v] - mn);
      p1[v] = __expf(s1[v] - mn);
      float rs = p0[v] + p1[v];
      rs += __shfl_xor(rs, 1, 32);
      rs += __shfl_xor(rs, 2, 32);
      rs += __shfl_xor(rs, 4, 32);
      rs += __shfl_xor(rs, 8, 32);
      l_s[v] = l_s[v] * alpha + rs;
      m_s[v] = mn;
#pragma unroll
      for (int g = 0; g < 4; ++g) oacc[g][v] *= alpha;
    }

    // ---- reshape P: C layout -> A layout via LDS (intra-wave; DS is in-order,
    //      so a dscnt wait is the only synchronization needed) ----
#pragma unroll
    for (int v = 0; v < 8; ++v) {
      int m = v + (half << 3);
      sP[m * 32 + lm] = p0[v];
      sP[m * 32 + 16 + lm] = p1[v];
    }
    lds_wait();
    v16bf pa;
    {
      const float4* p4a = (const float4*)(sP + lm * 32 + half * 8);
      const float4* p4b = (const float4*)(sP + lm * 32 + 16 + half * 8);
      put4(pa, 0, p4a[0]);
      put4(pa, 4, p4a[1]);
      put4(pa, 8, p4b[0]);
      put4(pa, 12, p4b[1]);
    }

    // ---- O += P (16x32) x V (32x64) ----
#pragma unroll
    for (int g = 0; g < 4; ++g) {
      v16bf vb;
#pragma unroll
      for (int j = 0; j < 16; ++j) {
        int key = c * 32 + (half << 3) + (j & 7) + ((j >> 3) << 4);
        vb[j] = f2bf(V[((size_t)(b * SK + key) * Hd) + head * DHd + (g << 4) + lm]);
      }
      oacc[g] = wmma_bf16(pa, vb, oacc[g]);
    }
  }

  // ---- normalize and write ----
#pragma unroll
  for (int g = 0; g < 4; ++g) {
#pragma unroll
    for (int v = 0; v < 8; ++v) {
      int qrow = qt * 16 + v + (half << 3);
      O[((size_t)(b * SQ + qrow) * Hd) + head * DHd + (g << 4) + lm] =
          oacc[g][v] / l_s[v];
    }
  }
}

// ============================== host sequencing ==============================
extern "C" void kernel_launch(void* const* d_in, const int* in_sizes, int n_in,
                              void* d_out, int out_size, void* d_ws, size_t ws_size,
                              hipStream_t stream) {
  (void)in_sizes; (void)n_in; (void)out_size; (void)ws_size;
  const int* x_enc = (const int*)d_in[0];
  const int* x_dec = (const int*)d_in[1];
  const float* emb_enc = (const float*)d_in[2];
  const float* W_enc_lin = (const float*)d_in[3];
  const float* emb_dec = (const float*)d_in[4];
  const float* W_dec_lin = (const float*)d_in[5];
  const float* p_decoder = (const float*)d_in[6];
  const float* e_pos = (const float*)d_in[7];
  const float* p_e_q = (const float*)d_in[8];
  const float* p_e_k = (const float*)d_in[9];
  const float* p_e_v = (const float*)d_in[10];
  const float* p_e_c = (const float*)d_in[11];
  const float* p_e_ff1 = (const float*)d_in[12];
  const float* p_e_ff2 = (const float*)d_in[13];
  const float* b_e_ff1 = (const float*)d_in[14];
  const float* b_e_ff2 = (const float*)d_in[15];
  const float* e_o_bias = (const float*)d_in[16];
  const float* e_o_scale = (const float*)d_in[17];
  const float* b_e_bias_1 = (const float*)d_in[18];
  const float* b_e_bias_2 = (const float*)d_in[19];
  const float* b_e_scale_1 = (const float*)d_in[20];
  const float* b_e_scale_2 = (const float*)d_in[21];
  const float* d_pos = (const float*)d_in[22];
  const float* p_d_q = (const float*)d_in[23];
  const float* p_d_k = (const float*)d_in[24];
  const float* p_d_v = (const float*)d_in[25];
  const float* p_d_c = (const float*)d_in[26];
  const float* p_a_q = (const float*)d_in[27];
  const float* p_a_k = (const float*)d_in[28];
  const float* p_a_v = (const float*)d_in[29];
  const float* p_a_c = (const float*)d_in[30];
  const float* p_d_ff1 = (const float*)d_in[31];
  const float* p_d_ff2 = (const float*)d_in[32];
  const float* b_d_ff1 = (const float*)d_in[33];
  const float* b_d_ff2 = (const float*)d_in[34];
  const float* d_o_bias = (const float*)d_in[35];
  const float* d_o_scale = (const float*)d_in[36];
  const float* b_d_bias_1 = (const float*)d_in[37];
  const float* b_d_bias_2 = (const float*)d_in[38];
  const float* b_d_bias_3 = (const float*)d_in[39];
  const float* b_d_scale_1 = (const float*)d_in[40];
  const float* b_d_scale_2 = (const float*)d_in[41];
  const float* b_d_scale_3 = (const float*)d_in[42];

  // ---- workspace carve-up (floats) ----
  float* wsf = (float*)d_ws;
  size_t off = 0;
  auto alloc = [&](size_t n) { float* p = wsf + off; off += n; return p; };
  const size_t SH = (size_t)SEQL * Hd;
  float* XE  = alloc((size_t)BSr * Ed);
  float* EMB = alloc((size_t)BSr * Hd);
  float* Xb  = alloc((size_t)BSr * Hd);
  float* LIb = alloc((size_t)BSr * Hd);
  float* Qb  = alloc((size_t)BSr * Hd);
  float* Kb  = alloc((size_t)BSr * Hd);
  float* Vb  = alloc((size_t)BSr * Hd);
  float* Tb  = alloc((size_t)BSr * Hd);
  float* H1b = alloc((size_t)BSr * Hd);
  float* ENC = alloc((size_t)BSr * Hd);
  float* DEM = alloc((size_t)BSr * Hd);
  float* FFb = alloc((size_t)BSr * Fd);

  auto gemm = [&](const float* Ap, const float* Wt, float* Cp, int M, int N, int K,
                  const float* bias, int relu) {
    dim3 g(N / 64, M / 128);
    gemm_bf16_wmma_kernel<<<g, 256, 0, stream>>>(Ap, Wt, Cp, M, N, K, bias, relu);
  };
  auto lnadd = [&](const float* x, const float* res, const float* bias,
                   const float* scale, float* out) {
    ln_add_kernel<<<BSr, 256, 0, stream>>>(x, res, bias, scale, out);
  };
  auto addpos = [&](const float* x, const float* pos, float* out) {
    int total4 = BSr * Hd / 4;
    add_pos_kernel<<<total4 / 256, 256, 0, stream>>>(
        (const float4*)x, (const float4*)pos, (float4*)out, total4,
        (int)(SH / 4 - 1));
  };
  auto attn = [&](const float* Qp, const float* Kp, const float* Vp, float* Op,
                  int SQ, int SK, int causal) {
    dim3 g(SQ / 16, NHd, Bc);
    attn_fa_wmma_kernel<<<g, 32, 0, stream>>>(Qp, Kp, Vp, Op, SQ, SK, causal);
  };

  const size_t HH = (size_t)Hd * Hd, HF = (size_t)Hd * Fd;

  // ================= encoder =================
  gather_kernel<<<BSr, 32, 0, stream>>>(x_enc, (const float4*)emb_enc, (float4*)XE);
  gemm(XE, W_enc_lin, EMB, BSr, Hd, Ed, nullptr, 0);
  const float* xcur = EMB;
  for (int l = 0; l < Ld; ++l) {
    addpos(xcur, e_pos + (size_t)l * SH, LIb);
    gemm(LIb, p_e_q + l * HH, Qb, BSr, Hd, Hd, nullptr, 0);
    gemm(LIb, p_e_k + l * HH, Kb, BSr, Hd, Hd, nullptr, 0);
    gemm(LIb, p_e_v + l * HH, Vb, BSr, Hd, Hd, nullptr, 0);
    attn(Qb, Kb, Vb, Tb, SEQL, SEQL, 0);
    gemm(Tb, p_e_c + l * HH, Qb, BSr, Hd, Hd, nullptr, 0);
    lnadd(Qb, LIb, b_e_bias_1 + l * Hd, b_e_scale_1 + l * Hd, H1b);
    gemm(H1b, p_e_ff1 + l * HF, FFb, BSr, Fd, Hd, b_e_ff1 + l * Fd, 1);
    gemm(FFb, p_e_ff2 + l * HF, Tb, BSr, Hd, Fd, b_e_ff2 + l * Hd, 0);
    lnadd(Tb, H1b, b_e_bias_2 + l * Hd, b_e_scale_2 + l * Hd, Xb);
    xcur = Xb;
  }
  lnadd(xcur, EMB, e_o_bias, e_o_scale, ENC);

  // ================= decoder =================
  gather_kernel<<<BSr, 32, 0, stream>>>(x_dec, (const float4*)emb_dec, (float4*)XE);
  gemm(XE, W_dec_lin, DEM, BSr, Hd, Ed, nullptr, 0);
  xcur = DEM;
  for (int l = 0; l < Ld; ++l) {
    addpos(xcur, d_pos + (size_t)l * SH, LIb);
    // masked self-attention
    gemm(LIb, p_d_q + l * HH, Qb, BSr, Hd, Hd, nullptr, 0);
    gemm(LIb, p_d_k + l * HH, Kb, BSr, Hd, Hd, nullptr, 0);
    gemm(LIb, p_d_v + l * HH, Vb, BSr, Hd, Hd, nullptr, 0);
    attn(Qb, Kb, Vb, Tb, SEQL, SEQL, 1);
    gemm(Tb, p_d_c + l * HH, Qb, BSr, Hd, Hd, nullptr, 0);
    lnadd(Qb, LIb, b_d_bias_1 + l * Hd, b_d_scale_1 + l * Hd, H1b);
    // cross-attention (keys/values from encoder output)
    gemm(H1b, p_a_q + l * HH, Qb, BSr, Hd, Hd, nullptr, 0);
    gemm(ENC, p_a_k + l * HH, Kb, BSr, Hd, Hd, nullptr, 0);
    gemm(ENC, p_a_v + l * HH, Vb, BSr, Hd, Hd, nullptr, 0);
    attn(Qb, Kb, Vb, Tb, SEQL, SEQL, 0);
    gemm(Tb, p_a_c + l * HH, Qb, BSr, Hd, Hd, nullptr, 0);
    lnadd(Qb, H1b, b_d_bias_2 + l * Hd, b_d_scale_2 + l * Hd, LIb);  // LIb = h2
    // feed-forward
    gemm(LIb, p_d_ff1 + l * HF, FFb, BSr, Fd, Hd, b_d_ff1 + l * Fd, 1);
    gemm(FFb, p_d_ff2 + l * HF, Tb, BSr, Hd, Fd, b_d_ff2 + l * Hd, 0);
    lnadd(Tb, LIb, b_d_bias_3 + l * Hd, b_d_scale_3 + l * Hd, Xb);
    xcur = Xb;
  }
  lnadd(xcur, DEM, d_o_bias, d_o_scale, H1b);  // H1b = dec_out
  gemm(H1b, p_decoder, (float*)d_out, BSr, VDd, Hd, nullptr, 0);
}
